// Encoder_6863357739331
// MI455X (gfx1250) — matrix operand order repaired and checked by reading, not compiled, not run on hardware
//
#include <hip/hip_runtime.h>
#include <cstdint>

typedef __bf16 bf16_t;
typedef __attribute__((ext_vector_type(16))) __bf16 v16bf;
typedef __attribute__((ext_vector_type(8)))  float  v8f;

#define DM   1024
#define DFF  4096
#define NH   16
#define DH   64
#define SEQ  1024
#define BATCH 4
#define ROWS (BATCH*SEQ)    // 4096
#define DEPTH_N 6

union Frag { uint4 u[2]; v16bf v; };

__device__ __forceinline__ v16bf ld_frag(const bf16_t* p0, const bf16_t* p1) {
    Frag f;
    f.u[0] = *(const uint4*)p0;
    f.u[1] = *(const uint4*)p1;
    return f.v;
}

__device__ __forceinline__ v8f wmma_bf16(v16bf a, v16bf b, v8f c) {
    // D(16x16 f32) = A(16x32 bf16) x B(32x16 bf16) + C
    return __builtin_amdgcn_wmma_f32_16x16x32_bf16(false, a, false, b,
                                                   (short)0, c, false, false);
}

// Async DMA: copy 16B per lane from global memory directly into LDS.
// Tracked by ASYNCcnt (NOT by the barrier), so s_wait_asynccnt must precede
// any barrier that publishes the data.
__device__ __forceinline__ void async_ld_b128(unsigned lds_off, const void* gaddr) {
    asm volatile("global_load_async_to_lds_b128 %0, %1, off"
                 :: "v"(lds_off), "v"(gaddr) : "memory");
}
__device__ __forceinline__ void wait_async() {
    asm volatile("s_wait_asynccnt 0" ::: "memory");
}
// Wave-relative LDS byte offset of a __shared__ pointer (flat LDS aperture
// maps addr[31:0] directly to the LDS address).
__device__ __forceinline__ unsigned lds_off_of(const void* p) {
    return (unsigned)(uintptr_t)p;
}

// ---------------------------------------------------------------------------
// Weight convert f32[K,N] -> bf16[N,K] (transpose), LDS-tiled, grid.z = layer
// ---------------------------------------------------------------------------
__global__ __launch_bounds__(256)
void wcvt(const float* __restrict__ w, bf16_t* __restrict__ wt, int K, int N) {
    __shared__ float tile[32][33];
    const size_t off = (size_t)blockIdx.z * (size_t)K * (size_t)N;
    const int k0 = blockIdx.y * 32, n0 = blockIdx.x * 32;
    const int tx = threadIdx.x & 31, ty = threadIdx.x >> 5;  // 32 x 8
#pragma unroll
    for (int i = 0; i < 4; ++i)
        tile[ty + 8 * i][tx] = w[off + (size_t)(k0 + ty + 8 * i) * N + n0 + tx];
    __syncthreads();
#pragma unroll
    for (int i = 0; i < 4; ++i)
        wt[off + (size_t)(n0 + ty + 8 * i) * K + k0 + tx] =
            (bf16_t)tile[tx][ty + 8 * i];
}

// ---------------------------------------------------------------------------
// Dual LayerNorm: one pass over x produces bf16 xn1 (attn branch) and xn2 (ffn)
// ---------------------------------------------------------------------------
__global__ __launch_bounds__(256)
void ln_dual(const float* __restrict__ x,
             const float* __restrict__ g1, const float* __restrict__ b1,
             const float* __restrict__ g2, const float* __restrict__ b2,
             bf16_t* __restrict__ o1, bf16_t* __restrict__ o2) {
    __shared__ float red[256];
    const int row = blockIdx.x, t = threadIdx.x;
    const float* xr = x + (size_t)row * DM;
    float vv[4];
    float s = 0.f;
#pragma unroll
    for (int i = 0; i < 4; ++i) { vv[i] = xr[t + 256 * i]; s += vv[i]; }
    red[t] = s; __syncthreads();
    for (int o = 128; o > 0; o >>= 1) { if (t < o) red[t] += red[t + o]; __syncthreads(); }
    const float mean = red[0] * (1.f / 1024.f);
    __syncthreads();
    float vs = 0.f;
#pragma unroll
    for (int i = 0; i < 4; ++i) { float d = vv[i] - mean; vs += d * d; }
    red[t] = vs; __syncthreads();
    for (int o = 128; o > 0; o >>= 1) { if (t < o) red[t] += red[t + o]; __syncthreads(); }
    const float inv = rsqrtf(red[0] * (1.f / 1024.f) + 1e-5f);
#pragma unroll
    for (int i = 0; i < 4; ++i) {
        const int c = t + 256 * i;
        const float xn = (vv[i] - mean) * inv;
        o1[(size_t)row * DM + c] = (bf16_t)(xn * g1[c] + b1[c]);
        o2[(size_t)row * DM + c] = (bf16_t)(xn * g2[c] + b2[c]);
    }
}

// Final LayerNorm -> f32 out
__global__ __launch_bounds__(256)
void ln_final(const float* __restrict__ x, const float* __restrict__ g,
              const float* __restrict__ b, float* __restrict__ out) {
    __shared__ float red[256];
    const int row = blockIdx.x, t = threadIdx.x;
    const float* xr = x + (size_t)row * DM;
    float vv[4];
    float s = 0.f;
#pragma unroll
    for (int i = 0; i < 4; ++i) { vv[i] = xr[t + 256 * i]; s += vv[i]; }
    red[t] = s; __syncthreads();
    for (int o = 128; o > 0; o >>= 1) { if (t < o) red[t] += red[t + o]; __syncthreads(); }
    const float mean = red[0] * (1.f / 1024.f);
    __syncthreads();
    float vs = 0.f;
#pragma unroll
    for (int i = 0; i < 4; ++i) { float d = vv[i] - mean; vs += d * d; }
    red[t] = vs; __syncthreads();
    for (int o = 128; o > 0; o >>= 1) { if (t < o) red[t] += red[t + o]; __syncthreads(); }
    const float inv = rsqrtf(red[0] * (1.f / 1024.f) + 1e-5f);
#pragma unroll
    for (int i = 0; i < 4; ++i) {
        const int c = t + 256 * i;
        out[(size_t)row * DM + c] = (vv[i] - mean) * inv * g[c] + b[c];
    }
}

// ---------------------------------------------------------------------------
// bf16 WMMA GEMM: C[M,N] = (A[M,K] x Wt[N,K]^T + bias) * scale
// Block 256 threads (8 waves). Block tile 128x128; wave tile 32(M)x64(N).
// Double-buffered LDS; tiles staged with async global->LDS DMA (ASYNCcnt):
// issue DMA for tile k+1, compute tile k from LDS, s_wait_asynccnt + barrier.
// ---------------------------------------------------------------------------
template <bool RELU, bool RESID>
__global__ __launch_bounds__(256)
void gemm_bf16(const bf16_t* __restrict__ A, const bf16_t* __restrict__ Wt,
               const float* __restrict__ bias, bf16_t* __restrict__ outb,
               float* __restrict__ resid, float scale, int M, int N, int K) {
    constexpr int LS = 40;  // padded LDS row stride (bf16): 16B aligned, conflict-free
    __shared__ bf16_t As[2][128 * LS];
    __shared__ bf16_t Bs[2][128 * LS];

    const int m0 = blockIdx.y * 128, n0 = blockIdx.x * 128;
    const int tid = threadIdx.x, lane = tid & 31, wave = tid >> 5;
    const int lrow = lane & 15, lhalf = lane >> 4;
    const int mo = (wave & 3) * 32;   // 4 waves across M
    const int no = (wave >> 2) * 64;  // 2 waves across N

    const v8f vz = {0.f, 0.f, 0.f, 0.f, 0.f, 0.f, 0.f, 0.f};
    v8f acc[2][4];
#pragma unroll
    for (int a = 0; a < 2; ++a)
#pragma unroll
        for (int b = 0; b < 4; ++b) acc[a][b] = vz;

    // staging: 4 threads per row, 8 bf16 (16B) each; rows sr and sr+64
    const int sr = tid >> 2, sc = (tid & 3) * 8;
    const bf16_t* Ag0 = A + (size_t)(m0 + sr) * K + sc;
    const bf16_t* Ag1 = A + (size_t)(m0 + sr + 64) * K + sc;
    const bf16_t* Bg0 = Wt + (size_t)(n0 + sr) * K + sc;
    const bf16_t* Bg1 = Wt + (size_t)(n0 + sr + 64) * K + sc;
    const int w0 = sr * LS + sc, w1 = (sr + 64) * LS + sc;
    unsigned dA0[2], dA1[2], dB0[2], dB1[2];
#pragma unroll
    for (int bb = 0; bb < 2; ++bb) {
        dA0[bb] = lds_off_of(As[bb] + w0);
        dA1[bb] = lds_off_of(As[bb] + w1);
        dB0[bb] = lds_off_of(Bs[bb] + w0);
        dB1[bb] = lds_off_of(Bs[bb] + w1);
    }

    // fragment offsets (ISA 16-bit A / B operand layouts)
    const int aOff0 = lhalf * 8, aOff1 = 16 + lhalf * 8;   // A: K 0-7/16-23 | 8-15/24-31
    const int bOff0 = lhalf * 8, bOff1 = 16 + lhalf * 8;   // B: K 0-7/16-23 | 8-15/24-31

    // preload tile 0 via async DMA
    async_ld_b128(dA0[0], Ag0);
    async_ld_b128(dA1[0], Ag1);
    async_ld_b128(dB0[0], Bg0);
    async_ld_b128(dB1[0], Bg1);
    wait_async();
    __syncthreads();

    int cur = 0;
    for (int kk = 0; kk < K; kk += 32) {
        const bool more = (kk + 32) < K;
        if (more) {  // issue next tile's DMA early; lands in the idle buffer
            const int nxt = cur ^ 1;
            async_ld_b128(dA0[nxt], Ag0 + kk + 32);
            async_ld_b128(dA1[nxt], Ag1 + kk + 32);
            async_ld_b128(dB0[nxt], Bg0 + kk + 32);
            async_ld_b128(dB1[nxt], Bg1 + kk + 32);
        }
        const bf16_t* aR0 = As[cur] + (mo + lrow) * LS;
        const bf16_t* aR1 = aR0 + 16 * LS;
        const v16bf af0 = ld_frag(aR0 + aOff0, aR0 + aOff1);
        const v16bf af1 = ld_frag(aR1 + aOff0, aR1 + aOff1);
#pragma unroll
        for (int nb = 0; nb < 4; ++nb) {
            const bf16_t* br = Bs[cur] + (no + nb * 16 + lrow) * LS;
            const v16bf bfv = ld_frag(br + bOff0, br + bOff1);
            acc[0][nb] = wmma_bf16(af0, bfv, acc[0][nb]);
            acc[1][nb] = wmma_bf16(af1, bfv, acc[1][nb]);
        }
        if (more) {
            wait_async();     // DMA writes to LDS complete
            __syncthreads();  // publish to all waves
            cur ^= 1;
        }
    }

    // epilogue: D layout -> VGPR r holds row (lhalf*8 + r), col = lrow
#pragma unroll
    for (int am = 0; am < 2; ++am) {
        const int mbase = m0 + mo + am * 16 + lhalf * 8;
#pragma unroll
        for (int nb = 0; nb < 4; ++nb) {
            const int n = n0 + no + nb * 16 + lrow;
            const float bv = bias[n];
#pragma unroll
            for (int r = 0; r < 8; ++r) {
                float v = (acc[am][nb][r] + bv) * scale;
                if (RELU) v = fmaxf(v, 0.f);
                const size_t idx = (size_t)(mbase + r) * N + n;
                if (RESID) resid[idx] += v;
                else       outb[idx] = (bf16_t)v;
            }
        }
    }
}

// ---------------------------------------------------------------------------
// Flash attention: block = 4 waves, one (b, h, 64-row q-block) per block.
// Q is pre-scaled by 1/sqrt(d) in its projection epilogue.
// K tile staged with async global->LDS DMA; V transposed via DS stores.
// ---------------------------------------------------------------------------
__global__ __launch_bounds__(128, 2)
void attn_flash(const bf16_t* __restrict__ q, const bf16_t* __restrict__ k,
                const bf16_t* __restrict__ v, const int* __restrict__ mask,
                bf16_t* __restrict__ ctx) {
    constexpr int KS = 72;  // K tile row stride: 64 d + pad (144B, 16B aligned)
    constexpr int VS = 40;  // Vt row stride: 32 keys + pad
    constexpr int PS = 40;  // P bounce row stride: 32 k + pad
    __shared__ bf16_t Kl[32 * KS];        // [key][d]
    __shared__ bf16_t Vt[64 * VS];        // [d][key] (transposed)
    __shared__ bf16_t Pb[4 * 16 * PS];    // per-wave P bounce

    const int b = blockIdx.z, h = blockIdx.y, q0 = blockIdx.x * 64;
    const int tid = threadIdx.x, lane = tid & 31, wave = tid >> 5;
    const int lrow = lane & 15, lhalf = lane >> 4;
    const int aOff0 = lhalf * 8, aOff1 = 16 + lhalf * 8;
    const int bOff0 = lhalf * 8, bOff1 = 16 + lhalf * 8;
    const size_t bS = (size_t)b * SEQ;

    // Q fragments (A layout): row = q0 + wave*16 + lrow, d split per lane half
    const bf16_t* qp = q + (bS + q0 + wave * 16 + lrow) * DM + h * DH;
    const v16bf qa0 = ld_frag(qp + aOff0,      qp + aOff1);       // d 0..31
    const v16bf qa1 = ld_frag(qp + 32 + aOff0, qp + 32 + aOff1);  // d 32..63

    const v8f vz = {0.f, 0.f, 0.f, 0.f, 0.f, 0.f, 0.f, 0.f};
    v8f o[4];
#pragma unroll
    for (int f = 0; f < 4; ++f) o[f] = vz;
    float mrow[8], lsum[8];
#pragma unroll
    for (int r = 0; r < 8; ++r) { mrow[r] = -3.0e38f; lsum[r] = 0.f; }

    const int qm = q0 + wave * 16 + lhalf * 8;  // D-layout row base
    const int* mbase = mask + ((size_t)b * SEQ + qm) * SEQ;
    bf16_t* Pw = Pb + wave * 16 * PS;

    // staging: 4 threads per key, 16 bf16 each
    const int skey = tid >> 2, sc = (tid & 3) * 16;
    const bf16_t* kstg = k + (bS + skey) * DM + h * DH + sc;
    const bf16_t* vstg = v + (bS + skey) * DM + h * DH + sc;
    const unsigned dK0 = lds_off_of(Kl + skey * KS + sc);
    const unsigned dK1 = lds_off_of(Kl + skey * KS + sc + 8);

    for (int j0 = 0; j0 < SEQ; j0 += 32) {
        __syncthreads();
        const size_t joff = (size_t)j0 * DM;
        async_ld_b128(dK0, kstg + joff);          // K tile: DMA straight to LDS
        async_ld_b128(dK1, kstg + joff + 8);
        {
            const bf16_t* vp = vstg + joff;       // V tile: transpose via DS
#pragma unroll
            for (int i = 0; i < 16; ++i) Vt[(sc + i) * VS + skey] = vp[i];
        }
        wait_async();
        __syncthreads();

        // scores for 32 keys: two 16x16 sub-tiles, each 2 WMMAs over d
        v8f s0 = vz, s1 = vz;
        {
            const bf16_t* kr0 = Kl + lrow * KS;          // keys j0..j0+15
            s0 = wmma_bf16(qa0, ld_frag(kr0 + bOff0,      kr0 + bOff1),      s0);
            s0 = wmma_bf16(qa1, ld_frag(kr0 + 32 + bOff0, kr0 + 32 + bOff1), s0);
            const bf16_t* kr1 = Kl + (16 + lrow) * KS;   // keys j0+16..j0+31
            s1 = wmma_bf16(qa0, ld_frag(kr1 + bOff0,      kr1 + bOff1),      s1);
            s1 = wmma_bf16(qa1, ld_frag(kr1 + 32 + bOff0, kr1 + 32 + bOff1), s1);
        }
        // mask (Q already carries the 1/sqrt(d) scale)
#pragma unroll
        for (int r = 0; r < 8; ++r) {
            const int mv0 = mbase[(size_t)r * SEQ + j0 + lrow];
            const int mv1 = mbase[(size_t)r * SEQ + j0 + 16 + lrow];
            const float a0 = s0[r], a1 = s1[r];
            s0[r] = mv0 ? a0 : -1.0e9f;
            s1[r] = mv1 ? a1 : -1.0e9f;
        }
        // online softmax (row reductions across the 16-lane half)
#pragma unroll
        for (int r = 0; r < 8; ++r) {
            float t = fmaxf(s0[r], s1[r]);
            t = fmaxf(t, __shfl_xor(t, 1));
            t = fmaxf(t, __shfl_xor(t, 2));
            t = fmaxf(t, __shfl_xor(t, 4));
            t = fmaxf(t, __shfl_xor(t, 8));
            const float mn = fmaxf(mrow[r], t);
            const float alpha = __expf(mrow[r] - mn);
            mrow[r] = mn;
            const float p0 = __expf(s0[r] - mn);
            const float p1 = __expf(s1[r] - mn);
            float ps = p0 + p1;
            ps += __shfl_xor(ps, 1);
            ps += __shfl_xor(ps, 2);
            ps += __shfl_xor(ps, 4);
            ps += __shfl_xor(ps, 8);
            lsum[r] = lsum[r] * alpha + ps;
#pragma unroll
            for (int f = 0; f < 4; ++f) o[f][r] *= alpha;
            // bounce P to LDS in [row M][k] form (wave-private; DS in-order per wave)
            Pw[(lhalf * 8 + r) * PS + lrow]      = (bf16_t)p0;
            Pw[(lhalf * 8 + r) * PS + 16 + lrow] = (bf16_t)p1;
        }
        // P (16x32) x V (32x64): read P back in A layout, V from transposed tile
        {
            const bf16_t* pr = Pw + lrow * PS;
            const v16bf pa = ld_frag(pr + aOff0, pr + aOff1);
#pragma unroll
            for (int f = 0; f < 4; ++f) {
                const bf16_t* vr = Vt + (f * 16 + lrow) * VS;
                o[f] = wmma_bf16(pa, ld_frag(vr + bOff0, vr + bOff1), o[f]);
            }
        }
    }

    // normalize + write ctx bf16 [B,S,Dm]
#pragma unroll
    for (int f = 0; f < 4; ++f) {
        const int n = h * DH + f * 16 + lrow;
#pragma unroll
        for (int r = 0; r < 8; ++r) {
            const float val = o[f][r] / lsum[r];
            ctx[(bS + qm + r) * DM + n] = (bf16_t)val;
        }
    }
}

// ---------------------------------------------------------------------------
extern "C" void kernel_launch(void* const* d_in, const int* in_sizes, int n_in,
                              void* d_out, int out_size, void* d_ws, size_t ws_size,
                              hipStream_t stream) {
    const float* x    = (const float*)d_in[0];
    const int*   mask = (const int*)d_in[1];
    const float* wq = (const float*)d_in[2];
    const float* bq = (const float*)d_in[3];
    const float* wk = (const float*)d_in[4];
    const float* bk = (const float*)d_in[5];
    const float* wv = (const float*)d_in[6];
    const float* bv = (const float*)d_in[7];
    const float* wo = (const float*)d_in[8];
    const float* bo = (const float*)d_in[9];
    const float* w1 = (const float*)d_in[10];
    const float* b1 = (const float*)d_in[11];
    const float* w2 = (const float*)d_in[12];
    const float* b2 = (const float*)d_in[13];
    const float* ln1g = (const float*)d_in[14];
    const float* ln1b = (const float*)d_in[15];
    const float* ln2g = (const float*)d_in[16];
    const float* ln2b = (const float*)d_in[17];
    const float* lnfg = (const float*)d_in[18];
    const float* lnfb = (const float*)d_in[19];

    // workspace layout (bf16 region first, then f32 residual buffer)
    bf16_t* base = (bf16_t*)d_ws;
    size_t off = 0;
    auto take = [&](size_t n) { bf16_t* p = base + off; off += n; return p; };
    const size_t MM = (size_t)DM * DM;  // 1M
    bf16_t* wqt = take(DEPTH_N * MM);
    bf16_t* wkt = take(DEPTH_N * MM);
    bf16_t* wvt = take(DEPTH_N * MM);
    bf16_t* wot = take(DEPTH_N * MM);
    bf16_t* w1t = take(DEPTH_N * 4 * MM);   // [Dff, Dm]
    bf16_t* w2t = take(DEPTH_N * 4 * MM);   // [Dm, Dff]
    bf16_t* xn1 = take((size_t)ROWS * DM);
    bf16_t* xn2 = take((size_t)ROWS * DM);
    bf16_t* qb  = take((size_t)ROWS * DM);
    bf16_t* kb  = take((size_t)ROWS * DM);
    bf16_t* vb  = take((size_t)ROWS * DM);
    bf16_t* cb  = take((size_t)ROWS * DM);
    bf16_t* hb  = take((size_t)ROWS * DFF);
    float* xw = (float*)(base + off);
    const size_t need = off * sizeof(bf16_t) + (size_t)ROWS * DM * sizeof(float);
    if (need > ws_size) return;  // workspace too small; bail deterministically

    hipMemcpyAsync(xw, x, (size_t)ROWS * DM * sizeof(float),
                   hipMemcpyDeviceToDevice, stream);

    // per-launch weight convert+transpose to bf16 [N,K]
    wcvt<<<dim3(32, 32, DEPTH_N),  256, 0, stream>>>(wq, wqt, DM, DM);
    wcvt<<<dim3(32, 32, DEPTH_N),  256, 0, stream>>>(wk, wkt, DM, DM);
    wcvt<<<dim3(32, 32, DEPTH_N),  256, 0, stream>>>(wv, wvt, DM, DM);
    wcvt<<<dim3(32, 32, DEPTH_N),  256, 0, stream>>>(wo, wot, DM, DM);
    wcvt<<<dim3(128, 32, DEPTH_N), 256, 0, stream>>>(w1, w1t, DM, DFF);
    wcvt<<<dim3(32, 128, DEPTH_N), 256, 0, stream>>>(w2, w2t, DFF, DM);

    const float QSCALE = 0.125f;  // 1/sqrt(64), folded into Q projection

    for (int i = 0; i < DEPTH_N; ++i) {
        ln_dual<<<ROWS, 256, 0, stream>>>(xw, ln1g + i * DM, ln1b + i * DM,
                                          ln2g + i * DM, ln2b + i * DM, xn1, xn2);
        gemm_bf16<false, false><<<dim3(8, 32), 256, 0, stream>>>(
            xn1, wqt + i * MM, bq + i * DM, qb, nullptr, QSCALE, ROWS, DM, DM);
        gemm_bf16<false, false><<<dim3(8, 32), 256, 0, stream>>>(
            xn1, wkt + i * MM, bk + i * DM, kb, nullptr, 1.0f, ROWS, DM, DM);
        gemm_bf16<false, false><<<dim3(8, 32), 256, 0, stream>>>(
            xn1, wvt + i * MM, bv + i * DM, vb, nullptr, 1.0f, ROWS, DM, DM);
        attn_flash<<<dim3(SEQ / 64, NH, BATCH), 128, 0, stream>>>(qb, kb, vb, mask, cb);
        gemm_bf16<false, true><<<dim3(8, 32), 256, 0, stream>>>(
            cb, wot + i * MM, bo + i * DM, nullptr, xw, 1.0f, ROWS, DM, DM);
        gemm_bf16<true, false><<<dim3(32, 32), 256, 0, stream>>>(
            xn2, w1t + i * 4 * MM, b1 + i * DFF, hb, nullptr, 1.0f, ROWS, DFF, DM);
        gemm_bf16<false, true><<<dim3(8, 32), 256, 0, stream>>>(
            hb, w2t + i * 4 * MM, b2 + i * DM, nullptr, xw, 1.0f, ROWS, DM, DFF);
    }
    ln_final<<<ROWS, 256, 0, stream>>>(xw, lnfg, lnfb, (float*)d_out);
}
